// NMT_74225624810060
// MI455X (gfx1250) — matrix operand order repaired and checked by reading, not compiled, not run on hardware
//
#include <hip/hip_runtime.h>
#include <hip/hip_bf16.h>
#include <math.h>

// ---------------------------------------------------------------------------
// Problem constants (from the reference)
// ---------------------------------------------------------------------------
#define S_   1024
#define T_   64          // decoder steps = T1-1
#define B_   32
#define E_   512
#define H_   512
#define V_   32000
#define G4_  2048        // 4*H
#define DD_  32          // attention window D

typedef __attribute__((ext_vector_type(16))) __bf16 v16bf;
typedef __attribute__((ext_vector_type(8)))  float  v8f;

__device__ __forceinline__ float sigm(float x) { return 1.0f / (1.0f + expf(-x)); }

// ---------------------------------------------------------------------------
// WMMA helpers (CDNA5 16x16x32 bf16, f32 accumulate)
// A: 16x32 bf16  — lane<16 holds row (lane), K 0..15; lane>=16 row lane-16, K 16..31
// B: 32x16 bf16  — lane<16 holds col (lane), K 0..15; lane>=16 col lane-16, K 16..31
// C: 16x16 f32   — lane<16 rows 0..7 (vgpr r -> row r), lane>=16 rows 8..15; col = lane&15
// ---------------------------------------------------------------------------
__device__ __forceinline__ v16bf load_a_frag(const __bf16* A, int lda, int m0, int k0, int lane) {
    int row = m0 + (lane & 15);
    int kb  = (lane >> 4) << 4;                 // 0 or 16
    const __bf16* p = A + (long)row * lda + k0 + kb;
    v16bf a;
#pragma unroll
    for (int j = 0; j < 16; ++j) a[j] = p[j];
    return a;
}

// B fragment from a row-major weight matrix W[N][K] (i.e. B = W^T)
__device__ __forceinline__ v16bf load_b_frag(const __bf16* W, int ldw, int n0, int k0, int lane) {
    int col = n0 + (lane & 15);
    int kb  = (lane >> 4) << 4;                 // 0 or 16
    const __bf16* p = W + (long)col * ldw + k0 + kb;
    v16bf b;
#pragma unroll
    for (int j = 0; j < 16; ++j) b[j] = p[j];
    return b;
}

__device__ __forceinline__ v8f wmma_bf16(v16bf a, v16bf b, v8f c) {
    return __builtin_amdgcn_wmma_f32_16x16x32_bf16(false, a, false, b, (short)0, c, false, false);
}

// ---------------------------------------------------------------------------
// Utility kernels
// ---------------------------------------------------------------------------
__global__ void k_cvt_bf16(const float* __restrict__ s, __bf16* __restrict__ d, long n) {
    long i = (long)blockIdx.x * blockDim.x + threadIdx.x;
    long stride = (long)gridDim.x * blockDim.x;
    for (; i < n; i += stride) d[i] = (__bf16)s[i];
}

// out[r][e] = bf16(emb[tok[r]][e])
__global__ void k_embed(const int* __restrict__ tok, const float* __restrict__ emb,
                        __bf16* __restrict__ out, long rows) {
    long i = (long)blockIdx.x * blockDim.x + threadIdx.x;
    long n = rows * E_;
    if (i >= n) return;
    long r = i >> 9;            // /512
    int  e = (int)(i & 511);
    out[i] = (__bf16)emb[(long)tok[r] * E_ + e];
}

__global__ void k_zero_state(float* ht, __bf16* htb) {
    int i = blockIdx.x * blockDim.x + threadIdx.x;
    if (i < B_ * H_) { ht[i] = 0.0f; htb[i] = (__bf16)0.0f; }
}

// ---------------------------------------------------------------------------
// preG = X @ enc_Wih^T + enc_b   (M=32768, N=2048, K=512), chip-wide WMMA GEMM
// grid: (N/128, M/16), block 256 (8 waves, one 16x16 tile each)
// ---------------------------------------------------------------------------
__global__ void __launch_bounds__(256)
k_gemm_bias(const __bf16* __restrict__ A, const __bf16* __restrict__ W,
            const float* __restrict__ bias, float* __restrict__ C, int M, int N, int K) {
    int lane = threadIdx.x & 31, wave = threadIdx.x >> 5;
    int m0 = blockIdx.y * 16;
    int n0 = (blockIdx.x * 8 + wave) * 16;
    v8f acc = {};
    for (int k0 = 0; k0 < K; k0 += 32) {
        __builtin_prefetch(W + (long)(n0 + (lane & 15)) * K + k0 + 64, 0, 0);
        v16bf a = load_a_frag(A, K, m0, k0, lane);
        v16bf b = load_b_frag(W, K, n0, k0, lane);
        acc = wmma_bf16(a, b, acc);
    }
    int col = n0 + (lane & 15);
    int rb  = (lane >> 4) * 8;
    float bv = bias[col];
#pragma unroll
    for (int r = 0; r < 8; ++r)
        C[(long)(m0 + rb + r) * N + col] = acc[r] + bv;
}

// ---------------------------------------------------------------------------
// Sequential encoder LSTM: single 1024-thread workgroup (32 waves), 1024 steps.
// Wave w owns gate columns [16w,16w+16) of all 4 gates (ntiles {w,w+32,w+64,w+96}),
// so the elementwise LSTM update is wave-local. h (bf16) and c (f32) live in LDS.
// ---------------------------------------------------------------------------
__global__ void __launch_bounds__(1024)
k_enc_seq(const float* __restrict__ preG,       // [S*B][2048] (bias folded in)
          const __bf16* __restrict__ Whh,       // [2048][512] bf16
          __bf16* __restrict__ enc_h,           // [B][S][H] bf16
          float* __restrict__ dec_h, __bf16* __restrict__ dec_hb,
          float* __restrict__ dec_c) {
    __shared__ __bf16 hb[B_][H_];
    __shared__ float  cs[B_][H_];
    int tid = threadIdx.x, lane = tid & 31, wave = tid >> 5;

    for (int i = tid; i < B_ * H_; i += 1024) {
        (&hb[0][0])[i] = (__bf16)0.0f;
        (&cs[0][0])[i] = 0.0f;
    }
    __syncthreads();

    int col = wave * 16 + (lane & 15);
    int rb  = (lane >> 4) * 8;

    for (int s = 0; s < S_; ++s) {
        v8f acc[2][4] = {};
        for (int k0 = 0; k0 < H_; k0 += 32) {
            v16bf a0 = load_a_frag(&hb[0][0], H_, 0, k0, lane);
            v16bf a1 = load_a_frag(&hb[0][0], H_, 16, k0, lane);
#pragma unroll
            for (int g = 0; g < 4; ++g) {
                int n0 = (g * 32 + wave) * 16;
                v16bf b = load_b_frag(Whh, H_, n0, k0, lane);
                acc[0][g] = wmma_bf16(a0, b, acc[0][g]);
                acc[1][g] = wmma_bf16(a1, b, acc[1][g]);
            }
        }
        __syncthreads();   // all reads of hb done before overwrite

        const float* pg = preG + (long)s * (B_ * G4_);
#pragma unroll
        for (int half = 0; half < 2; ++half) {
#pragma unroll
            for (int r = 0; r < 8; ++r) {
                int m = half * 16 + rb + r;          // batch index
                const float* row = pg + (long)m * G4_;
                float gi = acc[half][0][r] + row[col];
                float gf = acc[half][1][r] + row[512 + col];
                float gg = acc[half][2][r] + row[1024 + col];
                float go = acc[half][3][r] + row[1536 + col];
                float iv = sigm(gi), fv = sigm(gf), gv = tanhf(gg), ov = sigm(go);
                float c  = fv * cs[m][col] + iv * gv;
                float h  = ov * tanhf(c);
                cs[m][col] = c;
                hb[m][col] = (__bf16)h;
                enc_h[((long)m * S_ + s) * H_ + col] = (__bf16)h;
                if (s == S_ - 1) {
                    dec_h[m * H_ + col]  = h;
                    dec_hb[m * H_ + col] = (__bf16)h;
                    dec_c[m * H_ + col]  = c;
                }
            }
        }
        __syncthreads();
    }
}

// ---------------------------------------------------------------------------
// Decoder LSTM cell + p_t head. Single 1024-thread workgroup per step.
// x_t = [y_t (512), h~_t (512)]; g = x_t @ Wih^T + h @ Whh^T + b
// ---------------------------------------------------------------------------
__global__ void __launch_bounds__(1024)
k_dec_cell(const __bf16* __restrict__ ybf,        // [T][B][E]
           int t,
           const __bf16* __restrict__ ht_bf,      // h~ from previous step
           __bf16* __restrict__ dec_hb, float* __restrict__ dec_h,
           float* __restrict__ dec_c,
           const __bf16* __restrict__ Wih,        // [2048][1024]
           const __bf16* __restrict__ Whh,        // [2048][512]
           const float* __restrict__ bias,        // [2048]
           const float* __restrict__ Wtan,        // [512][512] f32
           const float* __restrict__ Wpt,         // [1][512]  f32
           float* __restrict__ center, int* __restrict__ left, int* __restrict__ right) {
    __shared__ union {
        __bf16 x[B_][2 * H_];      // GEMM input (phase 1)
        float  ytf[B_][H_];        // new h in f32 (phase 2, after GEMM)
    } u;
    __shared__ __bf16 hbl[B_][H_];

    int tid = threadIdx.x, lane = tid & 31, wave = tid >> 5;

    for (int i = tid; i < B_ * H_; i += 1024) {
        int bb = i >> 9, e = i & 511;
        u.x[bb][e]       = ybf[((long)t * B_ + bb) * E_ + e];
        u.x[bb][512 + e] = ht_bf[i];
        hbl[bb][e]       = dec_hb[i];
    }
    __syncthreads();

    v8f acc[2][4] = {};
    for (int k0 = 0; k0 < 2 * H_; k0 += 32) {       // input part, K=1024
        v16bf a0 = load_a_frag(&u.x[0][0], 2 * H_, 0, k0, lane);
        v16bf a1 = load_a_frag(&u.x[0][0], 2 * H_, 16, k0, lane);
#pragma unroll
        for (int g = 0; g < 4; ++g) {
            int n0 = (g * 32 + wave) * 16;
            v16bf b = load_b_frag(Wih, 2 * H_, n0, k0, lane);
            acc[0][g] = wmma_bf16(a0, b, acc[0][g]);
            acc[1][g] = wmma_bf16(a1, b, acc[1][g]);
        }
    }
    for (int k0 = 0; k0 < H_; k0 += 32) {           // recurrent part, K=512
        v16bf a0 = load_a_frag(&hbl[0][0], H_, 0, k0, lane);
        v16bf a1 = load_a_frag(&hbl[0][0], H_, 16, k0, lane);
#pragma unroll
        for (int g = 0; g < 4; ++g) {
            int n0 = (g * 32 + wave) * 16;
            v16bf b = load_b_frag(Whh, H_, n0, k0, lane);
            acc[0][g] = wmma_bf16(a0, b, acc[0][g]);
            acc[1][g] = wmma_bf16(a1, b, acc[1][g]);
        }
    }
    __syncthreads();   // GEMM done; safe to overwrite union with ytf

    int col = wave * 16 + (lane & 15);
    int rb  = (lane >> 4) * 8;
#pragma unroll
    for (int half = 0; half < 2; ++half) {
#pragma unroll
        for (int r = 0; r < 8; ++r) {
            int m = half * 16 + rb + r;
            float gi = acc[half][0][r] + bias[col];
            float gf = acc[half][1][r] + bias[512 + col];
            float gg = acc[half][2][r] + bias[1024 + col];
            float go = acc[half][3][r] + bias[1536 + col];
            float iv = sigm(gi), fv = sigm(gf), gv = tanhf(gg), ov = sigm(go);
            float c  = fv * dec_c[m * H_ + col] + iv * gv;
            float h  = ov * tanhf(c);
            dec_c[m * H_ + col]  = c;
            dec_h[m * H_ + col]  = h;
            dec_hb[m * H_ + col] = (__bf16)h;
            u.ytf[m][col]        = h;
        }
    }
    __syncthreads();

    // p_t head: wave `w` handles batch element w.
    // pt = sigmoid( sum_j Wpt[j] * tanh( yt . Wtan[j] ) )
    int bb = wave;
    float accp = 0.0f;
    for (int j = lane; j < H_; j += 32) {
        float d = 0.0f;
        const float* wr = Wtan + (long)j * H_;
        for (int h = 0; h < H_; ++h) d += u.ytf[bb][h] * wr[h];
        accp += tanhf(d) * Wpt[j];
    }
#pragma unroll
    for (int o = 16; o > 0; o >>= 1) accp += __shfl_down(accp, o, 32);
    if (lane == 0) {
        float pt  = sigm(accp);
        float ctr = (float)S_ * pt;
        int ci = (int)floorf(ctr);
        center[bb] = ctr;
        left[bb]   = max(0, ci - DD_);
        right[bb]  = min(S_, ci + DD_);
    }
}

// ---------------------------------------------------------------------------
// Local attention + context head, one block per batch element.
// ---------------------------------------------------------------------------
__global__ void __launch_bounds__(256)
k_dec_attn(const __bf16* __restrict__ enc_h,     // [B][S][H]
           const float* __restrict__ dec_h,      // yt f32
           const __bf16* __restrict__ dec_hb,    // yt bf16
           const float* __restrict__ center, const int* __restrict__ left,
           const int* __restrict__ right,
           const float* __restrict__ Wct2ht,     // [512][1024] f32
           float* __restrict__ ht, __bf16* __restrict__ ht_bf,
           __bf16* __restrict__ dec_out_bf, int t) {
    int b = blockIdx.x, tid = threadIdx.x;
    __shared__ __bf16 ytb[H_];
    __shared__ float  wbuf[S_];
    __shared__ float  red[256];
    __shared__ float  bct[2 * H_];

    for (int i = tid; i < H_; i += 256) ytb[i] = dec_hb[b * H_ + i];
    __syncthreads();

    float ctr = center[b];
    int L = left[b], R = right[b];

    float sc[4];
    float lmax = -3e38f;
#pragma unroll
    for (int q = 0; q < 4; ++q) {
        int s = q * 256 + tid;
        const __bf16* er = enc_h + ((long)b * S_ + s) * H_;
        float d = 0.0f;
        for (int h = 0; h < H_; ++h) d += (float)ytb[h] * (float)er[h];
        bool m = (s >= L) && (s < R);
        sc[q] = m ? d : -3e38f;
        lmax = fmaxf(lmax, sc[q]);
    }
    red[tid] = lmax; __syncthreads();
    for (int o = 128; o > 0; o >>= 1) { if (tid < o) red[tid] = fmaxf(red[tid], red[tid + o]); __syncthreads(); }
    float M = red[0]; __syncthreads();

    float lz = 0.0f;
#pragma unroll
    for (int q = 0; q < 4; ++q) {
        int s = q * 256 + tid;
        float w = 0.0f;
        if (sc[q] > -1e37f) {
            float e = expf(sc[q] - M);
            lz += e;
            float dp = (float)s - ctr;
            w = e * expf(-dp * dp / (DD_ * DD_ / 2.0f));
        }
        wbuf[s] = w;
    }
    red[tid] = lz; __syncthreads();
    for (int o = 128; o > 0; o >>= 1) { if (tid < o) red[tid] += red[tid + o]; __syncthreads(); }
    float invZ = 1.0f / red[0]; __syncthreads();

    // ct (only L..R contribute) and batch_ct = [ct, yt]
    for (int h = tid; h < H_; h += 256) {
        float a = 0.0f;
        for (int s = L; s < R; ++s)
            a += wbuf[s] * (float)enc_h[((long)b * S_ + s) * H_ + h];
        bct[h]       = a * invZ;
        bct[H_ + h]  = dec_h[b * H_ + h];
    }
    __syncthreads();

    // h~ = tanh(batch_ct @ Wct2ht^T)
    for (int j = tid; j < H_; j += 256) {
        float d = 0.0f;
        const float* wr = Wct2ht + (long)j * (2 * H_);
        for (int k = 0; k < 2 * H_; ++k) d += bct[k] * wr[k];
        float hv = tanhf(d);
        ht[b * H_ + j] = hv;
        __bf16 hvb = (__bf16)hv;
        ht_bf[b * H_ + j] = hvb;
        dec_out_bf[((long)t * B_ + b) * H_ + j] = hvb;
    }
}

// ---------------------------------------------------------------------------
// Final projection + fused online log-sum-exp + target gather.
// M = T*B = 2048 rows, N = 32000, K = 512. grid 128 blocks (16 rows each),
// 8 waves stride over the 2000 N-tiles, keeping running (max,sum) per row.
// ---------------------------------------------------------------------------
__global__ void __launch_bounds__(256)
k_final(const __bf16* __restrict__ A,            // dec_out bf16 [2048][512]
        const __bf16* __restrict__ Wf,           // [32000][512]
        const int* __restrict__ target,          // [T1][B]
        float* __restrict__ logp_tb) {
    __shared__ float tb[8][16][16];
    __shared__ float wmaxs[8][16], wsums[8][16], wtgts[8][16];
    int lane = threadIdx.x & 31, wave = threadIdx.x >> 5;
    int m0 = blockIdx.x * 16;

    int row = lane & 15;
    int m   = m0 + row;
    int tt  = m >> 5, bb = m & 31;
    int tgt = target[(tt + 1) * B_ + bb];

    float runmax = -3e38f, runsum = 0.0f, tlog = -3e38f;
    const int NT = V_ / 16;                      // 2000 tiles, divisible by 8
    for (int j = wave; j < NT; j += 8) {
        int n0 = j * 16;
        v8f acc = {};
        for (int k0 = 0; k0 < H_; k0 += 32) {
            __builtin_prefetch(Wf + (long)(n0 + 128 + (lane & 15)) * H_ + k0, 0, 0);
            v16bf a = load_a_frag(A, H_, m0, k0, lane);
            v16bf b = load_b_frag(Wf, H_, n0, k0, lane);
            acc = wmma_bf16(a, b, acc);
        }
        int c  = lane & 15;
        int rb = (lane >> 4) * 8;
#pragma unroll
        for (int r = 0; r < 8; ++r) tb[wave][rb + r][c] = acc[r];
        __syncthreads();
        if (lane < 16) {
            for (int c2 = 0; c2 < 16; ++c2) {
                float v  = tb[wave][row][c2];
                float nm = fmaxf(runmax, v);
                runsum = runsum * expf(runmax - nm) + expf(v - nm);
                runmax = nm;
            }
            if (tgt >= n0 && tgt < n0 + 16) tlog = tb[wave][row][tgt - n0];
        }
        __syncthreads();
    }
    if (lane < 16) {
        wmaxs[wave][row] = runmax;
        wsums[wave][row] = runsum;
        wtgts[wave][row] = tlog;
    }
    __syncthreads();
    if (wave == 0 && lane < 16) {
        float M = -3e38f;
#pragma unroll
        for (int w = 0; w < 8; ++w) M = fmaxf(M, wmaxs[w][row]);
        float Z = 0.0f;
#pragma unroll
        for (int w = 0; w < 8; ++w) Z += wsums[w][row] * expf(wmaxs[w][row] - M);
        float tg = -3e38f;
#pragma unroll
        for (int w = 0; w < 8; ++w) tg = fmaxf(tg, wtgts[w][row]);
        logp_tb[m] = (tgt != 0) ? (tg - M - logf(Z)) : 0.0f;   // PAD==0 mask
    }
}

__global__ void k_reduce(const float* __restrict__ logp_tb, float* __restrict__ out) {
    int b = threadIdx.x;
    if (b < B_) {
        float s = 0.0f;
        for (int t = 0; t < T_; ++t) s += logp_tb[t * B_ + b];
        out[b] = s;
    }
}

// ---------------------------------------------------------------------------
// Host-side orchestration
// ---------------------------------------------------------------------------
extern "C" void kernel_launch(void* const* d_in, const int* in_sizes, int n_in,
                              void* d_out, int out_size, void* d_ws, size_t ws_size,
                              hipStream_t stream) {
    const int*   source  = (const int*)  d_in[0];
    const int*   target  = (const int*)  d_in[1];
    const float* src_emb = (const float*)d_in[2];
    const float* tar_emb = (const float*)d_in[3];
    const float* enc_Wih = (const float*)d_in[4];
    const float* enc_Whh = (const float*)d_in[5];
    const float* enc_b   = (const float*)d_in[6];
    const float* dec_Wih = (const float*)d_in[7];
    const float* dec_Whh = (const float*)d_in[8];
    const float* dec_b   = (const float*)d_in[9];
    const float* W_tan   = (const float*)d_in[10];
    const float* W_pt    = (const float*)d_in[11];
    const float* W_ct2ht = (const float*)d_in[12];
    const float* W_fin   = (const float*)d_in[13];
    float* out = (float*)d_out;

    char* p = (char*)d_ws;
    auto alloc = [&](size_t bytes) -> char* {
        char* r = p;
        p += (bytes + 255) & ~(size_t)255;
        return r;
    };
    float*  preG    = (float*) alloc((size_t)S_ * B_ * G4_ * 4);   // 268 MB
    __bf16* xbf     = (__bf16*)alloc((size_t)S_ * B_ * E_ * 2);    // 33.5 MB
    __bf16* ybf     = (__bf16*)alloc((size_t)T_ * B_ * E_ * 2);    // 2 MB
    __bf16* ench    = (__bf16*)alloc((size_t)B_ * S_ * H_ * 2);    // 33.5 MB
    __bf16* WihE    = (__bf16*)alloc((size_t)G4_ * E_ * 2);
    __bf16* WhhE    = (__bf16*)alloc((size_t)G4_ * H_ * 2);
    __bf16* WihD    = (__bf16*)alloc((size_t)G4_ * (E_ + H_) * 2);
    __bf16* WhhD    = (__bf16*)alloc((size_t)G4_ * H_ * 2);
    __bf16* Wfin    = (__bf16*)alloc((size_t)V_ * H_ * 2);         // 32.8 MB
    float*  dec_h   = (float*) alloc((size_t)B_ * H_ * 4);
    __bf16* dec_hb  = (__bf16*)alloc((size_t)B_ * H_ * 2);
    float*  dec_c   = (float*) alloc((size_t)B_ * H_ * 4);
    float*  ht      = (float*) alloc((size_t)B_ * H_ * 4);
    __bf16* ht_bf   = (__bf16*)alloc((size_t)B_ * H_ * 2);
    __bf16* dec_obf = (__bf16*)alloc((size_t)T_ * B_ * H_ * 2);
    float*  centerb = (float*) alloc(B_ * 4);
    int*    leftb   = (int*)   alloc(B_ * 4);
    int*    rightb  = (int*)   alloc(B_ * 4);
    float*  logp_tb = (float*) alloc((size_t)T_ * B_ * 4);

    // 0) weight conversions to bf16 (all parallel)
    auto cvt = [&](const float* s, __bf16* d, long n) {
        int blocks = (int)((n + 255) / 256);
        k_cvt_bf16<<<blocks, 256, 0, stream>>>(s, d, n);
    };
    cvt(enc_Wih, WihE, (long)G4_ * E_);
    cvt(enc_Whh, WhhE, (long)G4_ * H_);
    cvt(dec_Wih, WihD, (long)G4_ * (E_ + H_));
    cvt(dec_Whh, WhhD, (long)G4_ * H_);
    cvt(W_fin,   Wfin, (long)V_ * H_);

    // 1) embedding gathers
    {
        long n = (long)S_ * B_ * E_;
        k_embed<<<(int)((n + 255) / 256), 256, 0, stream>>>(source, src_emb, xbf, (long)S_ * B_);
    }
    {
        long n = (long)T_ * B_ * E_;
        k_embed<<<(int)((n + 255) / 256), 256, 0, stream>>>(target, tar_emb, ybf, (long)T_ * B_);
    }

    // 2) preG = X @ enc_Wih^T + b  (chip-wide WMMA GEMM)
    {
        dim3 grid(G4_ / 128, (S_ * B_) / 16);
        k_gemm_bias<<<grid, 256, 0, stream>>>(xbf, WihE, enc_b, preG, S_ * B_, G4_, E_);
    }

    // 3) zero h~ state
    k_zero_state<<<(B_ * H_ + 255) / 256, 256, 0, stream>>>(ht, ht_bf);

    // 4) sequential encoder (persistent single workgroup, 32 waves)
    k_enc_seq<<<1, 1024, 0, stream>>>(preG, WhhE, ench, dec_h, dec_hb, dec_c);

    // 5) decoder: 64 sequential steps, 2 kernels per step
    for (int t = 0; t < T_; ++t) {
        k_dec_cell<<<1, 1024, 0, stream>>>(ybf, t, ht_bf, dec_hb, dec_h, dec_c,
                                           WihD, WhhD, dec_b, W_tan, W_pt,
                                           centerb, leftb, rightb);
        k_dec_attn<<<B_, 256, 0, stream>>>(ench, dec_h, dec_hb, centerb, leftb, rightb,
                                           W_ct2ht, ht, ht_bf, dec_obf, t);
    }

    // 6) final projection + fused log-softmax + gather
    k_final<<<(T_ * B_) / 16, 256, 0, stream>>>(dec_obf, Wfin, target, logp_tb);

    // 7) sum over time per batch element
    k_reduce<<<1, 64, 0, stream>>>(logp_tb, out);
}
